// _GCNNet_40286793236669
// MI455X (gfx1250) — matrix operand order repaired and checked by reading, not compiled, not run on hardware
//
#include <hip/hip_runtime.h>
#include <hip/hip_bf16.h>

typedef __attribute__((ext_vector_type(16))) _Float16 v16h;
typedef __attribute__((ext_vector_type(8)))  float    v8f;

#define NN 40000
#define EE 640000
#define DD 128
#define GG 64
#define LL 3
#define LDSW_STRIDE 144   // halves; byte stride 288 -> 32B-aligned, conflict-friendly

#if __has_builtin(__builtin_amdgcn_global_load_async_to_lds_b128)
#define HAVE_ASYNC_LDS 1
typedef int v4i_raw __attribute__((vector_size(16)));
typedef __attribute__((address_space(1))) v4i_raw* g_v4i;   // global v4i32*
typedef __attribute__((address_space(3))) v4i_raw* l_v4i;   // LDS    v4i32*
#else
#define HAVE_ASYNC_LDS 0
#endif

static __device__ __forceinline__ void wait_async_zero() {
#if __has_builtin(__builtin_amdgcn_s_wait_asynccnt)
    __builtin_amdgcn_s_wait_asynccnt(0);
#else
    asm volatile("s_wait_asynccnt 0x0" ::: "memory");
#endif
}

// ---------------- degree / normalization ----------------
__global__ void deg_init_kernel(float* deg, int n) {
    int i = blockIdx.x * blockDim.x + threadIdx.x;
    if (i < n) deg[i] = 1.0f;   // self loop contributes 1
}

__global__ void deg_acc_kernel(const int* __restrict__ dst, float* deg, int e) {
    int i = blockIdx.x * blockDim.x + threadIdx.x;
    if (i < e) atomicAdd(&deg[dst[i]], 1.0f);
}

__global__ void deg_inv_kernel(float* deg, int n) {
    int i = blockIdx.x * blockDim.x + threadIdx.x;
    if (i < n) {
        float d = deg[i];
        deg[i] = (d > 0.0f) ? rsqrtf(d) : 0.0f;   // in-place -> dinv
    }
}

// ---------------- f32 -> f16 convert ----------------
__global__ void to_f16_kernel(const float* __restrict__ in, _Float16* __restrict__ out, int n) {
    int i = blockIdx.x * blockDim.x + threadIdx.x;
    if (i < n) out[i] = (_Float16)in[i];
}

// ---------------- WMMA GEMM: out[N x 128] = H16[N x 128] @ W16[128 x 128] + b ----------------
// blockDim = 256 (8 waves). Each wave computes a 16 x 128 output panel.
// W16 is pre-converted f16 in global memory; staged into LDS via async DMA.
__global__ void gemm_wmma_kernel(const _Float16* __restrict__ H16,
                                 const _Float16* __restrict__ W16,
                                 const float* __restrict__ bias,
                                 float* __restrict__ out, int nrows) {
    __shared__ _Float16 ldsW[128 * LDSW_STRIDE];

    int t = threadIdx.x;
    // stage W16 (128x128 halves = 32KB) into LDS with padded row stride
#if HAVE_ASYNC_LDS
    // 2048 16-byte chunks, 8 per thread, via async global->LDS DMA (ASYNCcnt)
#pragma unroll
    for (int it = 0; it < 8; ++it) {
        int c = t + it * 256;            // chunk id: row k = c>>4, 8-half group = c&15
        int k = c >> 4, jc = c & 15;
        __builtin_amdgcn_global_load_async_to_lds_b128(
            (g_v4i)(W16 + (size_t)k * 128 + jc * 8),
            (l_v4i)(&ldsW[k * LDSW_STRIDE + jc * 8]),
            0, 0);
    }
    wait_async_zero();
#else
    for (int i = t; i < 128 * 128; i += 256) {
        int k = i >> 7, j = i & 127;
        ldsW[k * LDSW_STRIDE + j] = W16[i];
    }
#endif
    __syncthreads();

    int wave = t >> 5;
    int lane = t & 31;
    int rowbase = blockIdx.x * 128 + wave * 16;

    // ---- A fragments (16x32 f16 layout: lane = M, lane-half picks K group) ----
    int arow = rowbase + (lane & 15);
    if (arow >= nrows) arow = nrows - 1;            // clamp; stores are guarded
    const uint4* rowq = (const uint4*)(H16 + (size_t)arow * DD);  // 16 uint4 per row

    union AF { v16h v; uint4 q[2]; };
    AF a[4];
    int lh = lane >> 4;                              // lane half: 0 or 1
#pragma unroll
    for (int kt = 0; kt < 4; ++kt) {
        // pairs kt*16+pb+{0..3} -> uint4 idx kt*4+lh ; pairs +{8..11} -> idx kt*4+2+lh
        a[kt].q[0] = rowq[kt * 4 + lh];
        a[kt].q[1] = rowq[kt * 4 + 2 + lh];
    }

    int r0 = rowbase + (lh << 3);                    // C/D: lanes 16-31 hold M = r+8
    bool full = (rowbase + 16) <= nrows;

#pragma unroll
    for (int nt = 0; nt < 8; ++nt) {
        int col = nt * 16 + (lane & 15);
        float bv = bias[col];
        v8f c = {bv, bv, bv, bv, bv, bv, bv, bv};
#pragma unroll
        for (int kt = 0; kt < 4; ++kt) {
            // B 32x16 f16 layout: lane%16 = N, elements = K in same order as A
            v16h b;
#pragma unroll
            for (int s = 0; s < 16; ++s) {
                int k = kt * 32 + (s < 8 ? (lh << 3) + s : 16 + (lh << 3) + s - 8);
                b[s] = ldsW[k * LDSW_STRIDE + col];
            }
            c = __builtin_amdgcn_wmma_f32_16x16x32_f16(
                    false, a[kt].v, false, b, (short)0, c, false, false);
        }
        if (full) {
#pragma unroll
            for (int r = 0; r < 8; ++r)
                out[(size_t)(r0 + r) * DD + col] = c[r];
        } else {
#pragma unroll
            for (int r = 0; r < 8; ++r) {
                int row = r0 + r;
                if (row < nrows) out[(size_t)row * DD + col] = c[r];
            }
        }
    }
}

// ---------------- edge scatter-add aggregation ----------------
// one wave per edge; lane handles 4 features
__global__ void aggregate_kernel(const float* __restrict__ tin,
                                 const int* __restrict__ src,
                                 const int* __restrict__ dst,
                                 const float* __restrict__ dinv,
                                 float* __restrict__ outAgg,
                                 int nedge, int nnode) {
    long long gid = (long long)blockIdx.x * blockDim.x + threadIdx.x;
    int e = (int)(gid >> 5);
    int lane = (int)(gid & 31);
    if (e >= nedge + nnode) return;
    int s, d;
    if (e < nedge) { s = src[e]; d = dst[e]; }
    else           { s = e - nedge; d = s; }        // self loops
    float nrm = dinv[s] * dinv[d];
    const float4 v = *(const float4*)(tin + (size_t)s * DD + lane * 4);
    float* o = outAgg + (size_t)d * DD + lane * 4;
    atomicAdd(o + 0, nrm * v.x);
    atomicAdd(o + 1, nrm * v.y);
    atomicAdd(o + 2, nrm * v.z);
    atomicAdd(o + 3, nrm * v.w);
}

// ---------------- ReLU + BatchNorm statistics (per-column sum / sumsq of relu(h)) ----------------
__global__ void bn_stats_kernel(const float* __restrict__ h, float* __restrict__ sums, int n) {
    int c = threadIdx.x;                 // blockDim.x == 128
    float s = 0.0f, s2 = 0.0f;
    for (int r = blockIdx.x; r < n; r += gridDim.x) {
        float v = h[(size_t)r * DD + c];
        v = v > 0.0f ? v : 0.0f;
        s += v; s2 += v * v;
    }
    atomicAdd(&sums[c], s);
    atomicAdd(&sums[DD + c], s2);
}

__global__ void bn_apply_kernel(const float* __restrict__ h,
                                const float* __restrict__ sums,
                                const float* __restrict__ gamma,
                                const float* __restrict__ beta,
                                float* __restrict__ out, int n) {
    int i = blockIdx.x * blockDim.x + threadIdx.x;
    if (i >= n * DD) return;
    int c = i & (DD - 1);
    float invn = 1.0f / (float)n;
    float mean = sums[c] * invn;
    float var  = sums[DD + c] * invn - mean * mean;
    float v = h[i];
    v = v > 0.0f ? v : 0.0f;
    out[i] = gamma[c] * (v - mean) * rsqrtf(var + 1e-5f) + beta[c];
}

// ---------------- global mean pool ----------------
__global__ void pool_kernel(const float* __restrict__ h, const int* __restrict__ batch,
                            float* __restrict__ pooled, float* __restrict__ cnt, int n) {
    long long gid = (long long)blockIdx.x * blockDim.x + threadIdx.x;
    int node = (int)(gid >> 5);
    int lane = (int)(gid & 31);
    if (node >= n) return;
    int g = batch[node];
    const float4 v = *(const float4*)(h + (size_t)node * DD + lane * 4);
    float* o = pooled + (size_t)g * DD + lane * 4;
    atomicAdd(o + 0, v.x);
    atomicAdd(o + 1, v.y);
    atomicAdd(o + 2, v.z);
    atomicAdd(o + 3, v.w);
    if (lane == 0) atomicAdd(&cnt[g], 1.0f);
}

// ---------------- head MLP: relu(pooled/cnt @ w1 + b1) @ w2 + b2 ----------------
__global__ void mlp_kernel(const float* __restrict__ pooled, const float* __restrict__ cnt,
                           const float* __restrict__ w1, const float* __restrict__ b1,
                           const float* __restrict__ w2, const float* __restrict__ b2,
                           float* __restrict__ out) {
    __shared__ float red[64];
    int g = blockIdx.x, j = threadIdx.x;   // 64 blocks x 64 threads
    float inv = 1.0f / fmaxf(cnt[g], 1.0f);
    float s = b1[j];
#pragma unroll 8
    for (int k = 0; k < DD; ++k)
        s += pooled[(size_t)g * DD + k] * inv * w1[k * 64 + j];
    s = fmaxf(s, 0.0f);
    red[j] = s * w2[j];
    __syncthreads();
    for (int off = 32; off > 0; off >>= 1) {
        if (j < off) red[j] += red[j + off];
        __syncthreads();
    }
    if (j == 0) out[g] = red[0] + b2[0];
}

extern "C" void kernel_launch(void* const* d_in, const int* in_sizes, int n_in,
                              void* d_out, int out_size, void* d_ws, size_t ws_size,
                              hipStream_t stream) {
    const float* x      = (const float*)d_in[0];
    const int*   ei     = (const int*)d_in[1];
    const int*   batch  = (const int*)d_in[2];
    const float* Ws     = (const float*)d_in[3];
    const float* bs     = (const float*)d_in[4];
    const float* gammas = (const float*)d_in[5];
    const float* betas  = (const float*)d_in[6];
    const float* w1     = (const float*)d_in[7];
    const float* b1     = (const float*)d_in[8];
    const float* w2     = (const float*)d_in[9];
    const float* b2     = (const float*)d_in[10];
    float* out = (float*)d_out;

    const int* src = ei;           // edge_index[0]
    const int* dst = ei + EE;      // edge_index[1]

    // workspace layout (256B aligned slices)
    char* ws = (char*)d_ws;
    size_t off = 0;
    auto carve = [&](size_t bytes) -> char* {
        char* p = ws + off;
        off = (off + bytes + 255) & ~(size_t)255;
        return p;
    };
    float*    dinv    = (float*)   carve((size_t)NN * 4);
    _Float16* h16     = (_Float16*)carve((size_t)NN * DD * 2);
    _Float16* w16     = (_Float16*)carve((size_t)LL * DD * DD * 2);
    float*    bufT    = (float*)   carve((size_t)NN * DD * 4);
    float*    bufAgg  = (float*)   carve((size_t)NN * DD * 4);
    float*    bufH    = (float*)   carve((size_t)NN * DD * 4);
    float*    sums    = (float*)   carve(2 * DD * 4);
    float*    pooled  = (float*)   carve((GG * DD + GG) * 4);
    float*    cnt     = pooled + GG * DD;

    // degree -> dinv
    deg_init_kernel<<<(NN + 255) / 256, 256, 0, stream>>>(dinv, NN);
    deg_acc_kernel<<<(EE + 255) / 256, 256, 0, stream>>>(dst, dinv, EE);
    deg_inv_kernel<<<(NN + 255) / 256, 256, 0, stream>>>(dinv, NN);

    // convert all layer weights to f16 once
    to_f16_kernel<<<(LL * DD * DD + 255) / 256, 256, 0, stream>>>(Ws, w16, LL * DD * DD);

    const int nelem = NN * DD;
    const float* hin = x;
    for (int l = 0; l < LL; ++l) {
        to_f16_kernel<<<(nelem + 255) / 256, 256, 0, stream>>>(hin, h16, nelem);
        gemm_wmma_kernel<<<(NN + 127) / 128, 256, 0, stream>>>(
            h16, w16 + (size_t)l * DD * DD, bs + l * DD, bufT, NN);
        (void)hipMemsetAsync(bufAgg, 0, (size_t)nelem * 4, stream);
        {
            long long thr = (long long)(EE + NN) * 32;
            int blocks = (int)((thr + 255) / 256);
            aggregate_kernel<<<blocks, 256, 0, stream>>>(bufT, src, dst, dinv, bufAgg, EE, NN);
        }
        (void)hipMemsetAsync(sums, 0, 2 * DD * 4, stream);
        bn_stats_kernel<<<512, 128, 0, stream>>>(bufAgg, sums, NN);
        bn_apply_kernel<<<(nelem + 255) / 256, 256, 0, stream>>>(
            bufAgg, sums, gammas + l * DD, betas + l * DD, bufH, NN);
        hin = bufH;
    }

    (void)hipMemsetAsync(pooled, 0, (size_t)(GG * DD + GG) * 4, stream);
    {
        long long thr = (long long)NN * 32;
        int blocks = (int)((thr + 255) / 256);
        pool_kernel<<<blocks, 256, 0, stream>>>(bufH, batch, pooled, cnt, NN);
    }
    mlp_kernel<<<GG, 64, 0, stream>>>(pooled, cnt, w1, b1, w2, b2, out);
}
